// MDR_InteractingLayer_21208548508188
// MI455X (gfx1250) — matrix operand order repaired and checked by reading, not compiled, not run on hardware
//
#include <hip/hip_runtime.h>

// Problem constants (match reference)
#define BATCH 8192
#define FROWS 50
#define EMB   64
#define NHEAD 2
#define HDIM  32
#define NDOM  8

typedef __attribute__((ext_vector_type(16))) _Float16 v16h;
typedef __attribute__((ext_vector_type(8)))  float    v8f;

// LDS strides (elements). f16 stride 68 keeps 4B alignment for paired loads and
// spreads banks; f32 stride 66 avoids half-wave C-store conflicts.
#define F16LD 68
#define F32LD 66

// LDS layout (bytes). Bt (weights) is reused for S (scores) after projections;
// Outb reuses A+Q after the attention phase.
#define A_OFF    0                      // f16 [64][68]   = 8704
#define Q_OFF    8704                   // f16 [64][68]   = 8704
#define K_OFF    17408                  // f16 [64][68]   = 8704
#define VT_OFF   26112                  // f16 [64][68]   = 8704 (V transposed: [e][j])
#define P_OFF    34816                  // f16 [2][64][68]= 17408
#define RES_OFF  52224                  // f32 [64][66]   = 16896
#define BT_OFF   69120                  // f16 [256][68]  = 34816 (W^T, 4 mats)
#define S_OFF    BT_OFF                 // f32 [2][64][66]= 33792 (reuse of Bt)
#define OUTB_OFF 0                      // f32 [64][66]   = 16896 (reuse of A+Q)
#define SMEM_BYTES 103936

// ---- WMMA fragment loaders (contiguous k order, same in A and B, wave32) ----

// A-matrix 16x32 f16: lane l holds row m = l&15; elements 0..15 -> k = kb..kb+15,
// where kb = (l>=16 ? 16 : 0).
__device__ __forceinline__ v16h load_frag_a(const _Float16* buf, int ld,
                                            int m0, int k0, int lane) {
  int m  = m0 + (lane & 15);
  int kb = k0 + ((lane >> 4) << 4);
  const _Float16* p = buf + m * ld + kb;
  v16h r;
#pragma unroll
  for (int e = 0; e < 16; ++e) r[e] = p[e];
  return r;
}

// B-matrix 32x16 f16 (loaded from a B^T = [N][K] buffer): lane l holds col
// n = l&15; elements 0..15 -> k = kb..kb+15 with kb = (l>=16 ? 16 : 0).
__device__ __forceinline__ v16h load_frag_b(const _Float16* buf, int ld,
                                            int n0, int k0, int lane) {
  int n  = n0 + (lane & 15);
  int kb = k0 + ((lane >> 4) << 4);
  const _Float16* p = buf + n * ld + kb;
  v16h r;
#pragma unroll
  for (int e = 0; e < 16; ++e) r[e] = p[e];
  return r;
}

__device__ __forceinline__ v8f wmma_f16(v16h a, v16h b, v8f c) {
  // (neg_a, A, neg_b, B, c_mod, C, reuse_a, reuse_b)
  return __builtin_amdgcn_wmma_f32_16x16x32_f16(false, a, false, b,
                                                (short)0, c, false, false);
}

__global__ __launch_bounds__(256) void mdr_interacting_kernel(
    const float* __restrict__ inputs,       // [B,50,64]
    const int*   __restrict__ domain_ids,   // [B] (1..8)
    const float* __restrict__ Wq_dom,       // [8,64,64]
    const float* __restrict__ Wkey,         // [64,64]
    const float* __restrict__ Wval,         // [64,64]
    const float* __restrict__ Wres,         // [64,64]
    const float* __restrict__ ln_scale,     // [8,64]
    const float* __restrict__ ln_bias,      // [8,64]
    float*       __restrict__ out)          // [B,50,64]
{
  __shared__ __align__(16) char smem[SMEM_BYTES];
  __shared__ float s_mean[64];
  __shared__ float s_rstd[64];

  _Float16* Abuf  = (_Float16*)(smem + A_OFF);
  _Float16* Qbuf  = (_Float16*)(smem + Q_OFF);
  _Float16* Kbuf  = (_Float16*)(smem + K_OFF);
  _Float16* Vtbuf = (_Float16*)(smem + VT_OFF);
  _Float16* Pbuf  = (_Float16*)(smem + P_OFF);
  _Float16* Btbuf = (_Float16*)(smem + BT_OFF);
  float*    Resb  = (float*)(smem + RES_OFF);
  float*    Sbuf  = (float*)(smem + S_OFF);
  float*    Outb  = (float*)(smem + OUTB_OFF);

  const int b    = blockIdx.x;
  const int tid  = threadIdx.x;
  const int lane = tid & 31;
  // Scalarize the wave id: all tile coordinates derived from it become SGPRs,
  // so per-wave branches are s_cbranch (no EXEC divergence in epilogues).
  const int swave = __builtin_amdgcn_readfirstlane(tid >> 5);

  int idx = domain_ids[b] - 1;
  idx = (idx < 0) ? 0 : ((idx > NDOM - 1) ? NDOM - 1 : idx);
  idx = __builtin_amdgcn_readfirstlane(idx);   // block-uniform -> SGPR bases

  // ---------------- Phase 1: stage inputs (f16, padded) + W^T (f16) --------
  const float* inp = inputs + (size_t)b * (FROWS * EMB);
#pragma unroll 4
  for (int i = tid; i < 64 * 64; i += 256) {
    int r = i >> 6, c = i & 63;
    float v = (r < FROWS) ? inp[i] : 0.0f;   // i == r*64 + c
    Abuf[r * F16LD + c] = (_Float16)v;
  }
  {
    const float* srcs[4] = {Wq_dom + (size_t)idx * 4096, Wkey, Wval, Wres};
#pragma unroll
    for (int mat = 0; mat < 4; ++mat) {
      const float* src = srcs[mat];
#pragma unroll 4
      for (int r = tid; r < 4096; r += 256) {
        int k = r >> 6, n = r & 63;            // src row-major [k][n]
        Btbuf[(mat * 64 + n) * F16LD + k] = (_Float16)src[r];  // transposed
      }
    }
  }
  __syncthreads();

  // ---------------- Phase 2: fused projections X @ [Wq|Wk|Wv|Wres] ---------
  // Wave pair (swave>>1) owns one output matrix; each wave does 8 tiles
  // (4 M-tiles x 2 N-tiles), K = 64 -> 2 WMMA steps per tile.
  {
    const int mat  = swave >> 1;   // scalar: 0=Q 1=K 2=V 3=Res
    const int half = swave & 1;    // scalar: which pair of N-tiles
    const int mrow_off = (lane >> 4) << 3;
    const int nlane    = lane & 15;

    v16h afr[4][2];
#pragma unroll
    for (int mt = 0; mt < 4; ++mt) {
      afr[mt][0] = load_frag_a(Abuf, F16LD, mt * 16, 0,  lane);
      afr[mt][1] = load_frag_a(Abuf, F16LD, mt * 16, 32, lane);
    }

#pragma unroll
    for (int nn = 0; nn < 2; ++nn) {
      const int ncol = (half * 2 + nn) << 4;
      v16h b0 = load_frag_b(Btbuf, F16LD, mat * 64 + ncol, 0,  lane);
      v16h b1 = load_frag_b(Btbuf, F16LD, mat * 64 + ncol, 32, lane);
#pragma unroll
      for (int mt = 0; mt < 4; ++mt) {
        v8f c = {};
        c = wmma_f16(afr[mt][0], b0, c);
        c = wmma_f16(afr[mt][1], b1, c);
        const int mrow = mt * 16 + mrow_off;
        const int ncl  = ncol + nlane;
        if (mat == 0) {
#pragma unroll
          for (int r2 = 0; r2 < 8; ++r2)
            Qbuf[(mrow + r2) * F16LD + ncl] = (_Float16)c[r2];
        } else if (mat == 1) {
#pragma unroll
          for (int r2 = 0; r2 < 8; ++r2)
            Kbuf[(mrow + r2) * F16LD + ncl] = (_Float16)c[r2];
        } else if (mat == 2) {
#pragma unroll
          for (int r2 = 0; r2 < 8; ++r2)
            Vtbuf[ncl * F16LD + (mrow + r2)] = (_Float16)c[r2];  // transposed
        } else {
#pragma unroll
          for (int r2 = 0; r2 < 8; ++r2)
            Resb[(mrow + r2) * F32LD + ncl] = c[r2];
        }
      }
    }
  }
  __syncthreads();

  // ---------------- Phase 3: scores S_h = Q_h K_h^T / sqrt(HD) -------------
  // Wave -> (head = swave>>2, mt = swave&3); 4 j-tiles, K = HD = 32.
  {
    const int head = swave >> 2;
    const int mt   = swave & 3;
    v16h a = load_frag_a(Qbuf, F16LD, mt * 16, head * HDIM, lane);
    float* Sh = Sbuf + head * (64 * F32LD);
    const int mrow = mt * 16 + ((lane >> 4) << 3);
#pragma unroll
    for (int jt = 0; jt < 4; ++jt) {
      v16h bb = load_frag_b(Kbuf, F16LD, jt * 16, head * HDIM, lane);
      v8f c = {};
      c = wmma_f16(a, bb, c);
      const int j = jt * 16 + (lane & 15);
#pragma unroll
      for (int r2 = 0; r2 < 8; ++r2)
        Sh[(mrow + r2) * F32LD + j] = c[r2] * 0.17677669529663689f; // 1/sqrt(32)
    }
  }
  __syncthreads();

  // ---------------- Phase 4: masked softmax over j<50, write P (f16) -------
  if (tid < NHEAD * 64) {
    int head = tid >> 6, row = tid & 63;
    float*    Srow = Sbuf + head * (64 * F32LD) + row * F32LD;
    _Float16* Prow = Pbuf + head * (64 * F16LD) + row * F16LD;
    float mx = -1e30f;
    for (int j = 0; j < FROWS; ++j) mx = fmaxf(mx, Srow[j]);
    float s = 0.0f;
    for (int j = 0; j < FROWS; ++j) {
      float e = __expf(Srow[j] - mx);
      Srow[j] = e;
      s += e;
    }
    float inv = 1.0f / s;
    for (int j = 0; j < FROWS; ++j) Prow[j] = (_Float16)(Srow[j] * inv);
    for (int j = FROWS; j < 64; ++j) Prow[j] = (_Float16)0.0f;  // mask padding
  }
  __syncthreads();

  // ---------------- Phase 5: O_h = P_h V_h ; + Res ; ReLU -------------------
  // Wave -> (head = swave>>2, mt = swave&3); 2 N-tiles, K = 64 -> 2 steps.
  {
    const int head = swave >> 2;
    const int mt   = swave & 3;
    const _Float16* Ph = Pbuf + head * (64 * F16LD);
    v16h a0 = load_frag_a(Ph, F16LD, mt * 16, 0,  lane);
    v16h a1 = load_frag_a(Ph, F16LD, mt * 16, 32, lane);
    const int mrow = mt * 16 + ((lane >> 4) << 3);
#pragma unroll
    for (int nt2 = 0; nt2 < 2; ++nt2) {
      v16h b0 = load_frag_b(Vtbuf, F16LD, head * HDIM + nt2 * 16, 0,  lane);
      v16h b1 = load_frag_b(Vtbuf, F16LD, head * HDIM + nt2 * 16, 32, lane);
      v8f c = {};
      c = wmma_f16(a0, b0, c);
      c = wmma_f16(a1, b1, c);
      const int e = head * HDIM + nt2 * 16 + (lane & 15);
#pragma unroll
      for (int r2 = 0; r2 < 8; ++r2) {
        int m = mrow + r2;
        float v = c[r2] + Resb[m * F32LD + e];
        Outb[m * F32LD + e] = fmaxf(v, 0.0f);
      }
    }
  }
  __syncthreads();

  // ---------------- Phase 6a: per-row mean / rstd ---------------------------
  if (tid < FROWS) {
    const float* row = Outb + tid * F32LD;
    float mean = 0.0f;
#pragma unroll
    for (int e2 = 0; e2 < 64; ++e2) mean += row[e2];
    mean *= (1.0f / 64.0f);
    float var = 0.0f;
#pragma unroll
    for (int e2 = 0; e2 < 64; ++e2) { float d = row[e2] - mean; var += d * d; }
    var *= (1.0f / 64.0f);
    s_mean[tid] = mean;
    s_rstd[tid] = rsqrtf(var + 1e-6f);
  }
  __syncthreads();

  // ---------------- Phase 6b: LayerNorm affine + coalesced store -----------
  const float* g  = ln_scale + (size_t)idx * EMB;
  const float* be = ln_bias  + (size_t)idx * EMB;
  float* orow = out + (size_t)b * (FROWS * EMB);
#pragma unroll 4
  for (int i = tid; i < FROWS * EMB; i += 256) {
    int r = i >> 6, c = i & 63;
    float v = (Outb[r * F32LD + c] - s_mean[r]) * s_rstd[r];
    orow[i] = v * g[c] + be[c];
  }
}

extern "C" void kernel_launch(void* const* d_in, const int* in_sizes, int n_in,
                              void* d_out, int out_size, void* d_ws, size_t ws_size,
                              hipStream_t stream) {
  const float* inputs  = (const float*)d_in[0];
  const int*   dom     = (const int*)d_in[1];
  const float* Wq_dom  = (const float*)d_in[2];
  const float* Wkey    = (const float*)d_in[3];
  const float* Wval    = (const float*)d_in[4];
  const float* Wres    = (const float*)d_in[5];
  const float* lns     = (const float*)d_in[6];
  const float* lnb     = (const float*)d_in[7];
  float* out = (float*)d_out;
  (void)in_sizes; (void)n_in; (void)out_size; (void)d_ws; (void)ws_size;

  hipLaunchKernelGGL(mdr_interacting_kernel, dim3(BATCH), dim3(256), 0, stream,
                     inputs, dom, Wq_dom, Wkey, Wval, Wres, lns, lnb, out);
}
